// QuantizedHadamardLinear_6459630813323
// MI455X (gfx1250) — 8-bit matrix operand order repaired and checked by reading, compiled, not run on hardware
//
#include <hip/hip_runtime.h>
#include <hip/hip_bf16.h>
#include <stdint.h>

typedef int v8i __attribute__((ext_vector_type(8)));

#define IN_F   3072
#define PADDED 4096
#define OUT_F  4096

__global__ __launch_bounds__(256)
void fwht_quant_kernel(const float* __restrict__ in, int in_cols,
                       unsigned char* __restrict__ q_out,
                       int* __restrict__ rowsum) {
    __shared__ float buf[PADDED];
    __shared__ int   ssum[256];

    const int row = blockIdx.x;
    const float* src = in + (size_t)row * in_cols;

    for (int i = threadIdx.x; i < PADDED; i += 256)
        buf[i] = (i < in_cols) ? src[i] : 0.0f;
    __syncthreads();

    for (int stride = 1; stride < PADDED; stride <<= 1) {
        for (int t = threadIdx.x; t < PADDED / 2; t += 256) {
            int i = ((t / stride) * (stride << 1)) + (t % stride);
            float a = buf[i];
            float b = buf[i + stride];
            buf[i]          = a + b;
            buf[i + stride] = a - b;
        }
        __syncthreads();
    }

    const float inv_scale = 255.0f / 768.0f;
    unsigned int* qo = (unsigned int*)(q_out + (size_t)row * PADDED);
    int local = 0;
    for (int i4 = threadIdx.x; i4 < PADDED / 4; i4 += 256) {
        unsigned int pack = 0;
        #pragma unroll
        for (int j = 0; j < 4; ++j) {
            float v  = buf[i4 * 4 + j];
            v        = fminf(fmaxf(v, -384.0f), 384.0f);
            float qf = rintf((v + 384.0f) * inv_scale);
            qf       = fminf(fmaxf(qf, 0.0f), 255.0f);
            unsigned int q = (unsigned int)qf;
            local   += (int)q;
            pack    |= q << (8 * j);
        }
        qo[i4] = pack;
    }

    ssum[threadIdx.x] = local;
    __syncthreads();
    for (int off = 128; off > 0; off >>= 1) {
        if ((int)threadIdx.x < off) ssum[threadIdx.x] += ssum[threadIdx.x + off];
        __syncthreads();
    }
    if (threadIdx.x == 0) rowsum[row] = ssum[0];
}

__global__ __launch_bounds__(256)
void gemm_iu8_kernel(const unsigned char* __restrict__ A,
                     const unsigned char* __restrict__ B,
                     const int* __restrict__ rsA,
                     const int* __restrict__ rsB,
                     const float* __restrict__ bias,
                     float* __restrict__ C,
                     int M, int N) {
    const int K    = PADDED;
    const int lane = threadIdx.x & 31;
    const int wave = threadIdx.x >> 5;
    const int wm   = wave >> 2;
    const int wn   = wave & 3;
    const int l15  = lane & 15;
    const int half = lane >> 4;

    const int tileM = blockIdx.y * 128 + wm * 64;
    const int tileN = blockIdx.x * 128 + wn * 32;

    uint32_t aoff[4];
    #pragma unroll
    for (int tm = 0; tm < 4; ++tm)
        aoff[tm] = (uint32_t)(tileM + tm * 16 + l15) * K + half * 16;

    uint32_t boff[2];
    #pragma unroll
    for (int tn = 0; tn < 2; ++tn)
        boff[tn] = (uint32_t)(tileN + tn * 16 + l15) * K + half * 16;

    v8i acc[4][2];
    #pragma unroll
    for (int tm = 0; tm < 4; ++tm)
        #pragma unroll
        for (int tn = 0; tn < 2; ++tn)
            acc[tm][tn] = (v8i)(0);

    for (uint32_t k0 = 0; k0 < (uint32_t)K; k0 += 64) {
        v8i a[4], b[2];
        #pragma unroll
        for (int tm = 0; tm < 4; ++tm) {
            const unsigned char* p = A + (aoff[tm] + k0);
            uint2 g0 = *(const uint2*)(p);
            uint2 g1 = *(const uint2*)(p + 8);
            uint2 g2 = *(const uint2*)(p + 32);
            uint2 g3 = *(const uint2*)(p + 40);
            a[tm][0] = (int)g0.x; a[tm][1] = (int)g0.y;
            a[tm][2] = (int)g1.x; a[tm][3] = (int)g1.y;
            a[tm][4] = (int)g2.x; a[tm][5] = (int)g2.y;
            a[tm][6] = (int)g3.x; a[tm][7] = (int)g3.y;
        }
        #pragma unroll
        for (int tn = 0; tn < 2; ++tn) {
            const unsigned char* p = B + (boff[tn] + k0);
            uint4 lo = *(const uint4*)(p);
            uint4 hi = *(const uint4*)(p + 32);
            b[tn][0] = (int)lo.x; b[tn][1] = (int)lo.y;
            b[tn][2] = (int)lo.z; b[tn][3] = (int)lo.w;
            b[tn][4] = (int)hi.x; b[tn][5] = (int)hi.y;
            b[tn][6] = (int)hi.z; b[tn][7] = (int)hi.w;
        }

        #pragma unroll
        for (int tm = 0; tm < 4; ++tm)
            #pragma unroll
            for (int tn = 0; tn < 2; ++tn)
                acc[tm][tn] = __builtin_amdgcn_wmma_i32_16x16x64_iu8(
                     false, a[tm],
                     false, b[tn],
                    acc[tm][tn],
                     false,  false);
    }

    const float s    = 768.0f / 255.0f;
    const float s2   = s * s;
    const float smn  = s * (-384.0f);
    const float c0   = (float)PADDED * 384.0f * 384.0f;
    const float invP = 1.0f / (float)PADDED;

    #pragma unroll
    for (int tn = 0; tn < 2; ++tn) {
        const int   n  = tileN + tn * 16 + l15;
        const float bn = bias[n];
        const float rw = (float)rsB[n];
        #pragma unroll
        for (int tm = 0; tm < 4; ++tm) {
            const int mbase = tileM + tm * 16 + half * 8;
            #pragma unroll
            for (int r = 0; r < 8; ++r) {
                const int   m  = mbase + r;
                const float rx = (float)rsA[m];
                const float v  = (s2 * (float)acc[tm][tn][r]
                                  + smn * (rx + rw) + c0) * invP + bn;
                C[(size_t)m * N + n] = v;
            }
        }
    }
}

extern "C" void kernel_launch(void* const* d_in, const int* in_sizes, int n_in,
                              void* d_out, int out_size, void* d_ws, size_t ws_size,
                              hipStream_t stream) {
    const float* x      = (const float*)d_in[0];
    const float* weight = (const float*)d_in[1];
    const float* bias   = (const float*)d_in[2];
    float*       out    = (float*)d_out;

    const int rows_x = in_sizes[0] / IN_F;
    const int rows_w = in_sizes[1] / IN_F;

    unsigned char* ws  = (unsigned char*)d_ws;
    unsigned char* qx  = ws;
    unsigned char* qw  = qx + (size_t)rows_x * PADDED;
    int*           rsx = (int*)(qw + (size_t)rows_w * PADDED);
    int*           rsw = rsx + rows_x;

    fwht_quant_kernel<<<rows_x, 256, 0, stream>>>(x,      IN_F, qx, rsx);
    fwht_quant_kernel<<<rows_w, 256, 0, stream>>>(weight, IN_F, qw, rsw);

    dim3 grid(rows_w / 128, rows_x / 128);
    gemm_iu8_kernel<<<grid, 256, 0, stream>>>(qx, qw, rsx, rsw, bias, out,
                                              rows_x, rows_w);
}
